// SE3EquivariantLayer_83270825935544
// MI455X (gfx1250) — matrix operand order repaired and checked by reading, not compiled, not run on hardware
//
#include <hip/hip_runtime.h>
#include <hip/hip_bf16.h>

// ---------------------------------------------------------------------------
// EGNN / SE3-equivariant layer, fully fused, bf16 WMMA (gfx1250 wave32).
// ---------------------------------------------------------------------------

typedef __attribute__((ext_vector_type(16))) __bf16 v16bf;
typedef __attribute__((ext_vector_type(8)))  float  v8f;

union ABFrag { uint4 q[2]; v16bf v; };

__device__ __forceinline__ unsigned short f2bf(float f) {
  unsigned int u = __float_as_uint(f);
  unsigned int r = u + 0x7FFFu + ((u >> 16) & 1u);   // round-to-nearest-even
  return (unsigned short)(r >> 16);
}

// Fast SiLU: x * rcp(1+exp(-x)) -> v_exp_f32 + v_rcp_f32 + v_mul (no precise-div
// expansion with v_div_scale/v_div_fixup chains).
__device__ __forceinline__ float silu_f(float x) {
  float d = 1.0f + __expf(-x);
  return x * __builtin_amdgcn_rcpf(d);
}

// ---------------------------------------------------------------------------
// Pack a row-major [Ktot x 128] fp32 weight matrix into WMMA B-fragments
// (bf16). Fragment f = kt*8 + nt covers K rows [kt*32, kt*32+32), N cols
// [nt*16, nt*16+16). Per-lane layout (16x16x32 bf16 B-matrix):
//   lane l, element e:  K = kt*32 + (e<8 ? e : e+8) + (l>=16 ? 8 : 0),  N = nt*16 + (l&15)
// ---------------------------------------------------------------------------
__global__ void pack_wmma_b(const float* __restrict__ W, int Ktot,
                            unsigned short* __restrict__ dst) {
  int total = (Ktot >> 5) * 8 * 512;
  int idx = blockIdx.x * 256 + threadIdx.x;
  if (idx >= total) return;
  int e    = idx & 15;
  int lane = (idx >> 4) & 31;
  int f    = idx >> 9;
  int kt   = f >> 3;
  int nt   = f & 7;
  int K = kt * 32 + ((e < 8) ? e : e + 8) + ((lane >= 16) ? 8 : 0);
  int n = nt * 16 + (lane & 15);
  dst[idx] = f2bf(W[(size_t)K * 128 + n]);
}

__global__ void zero_f32(float* __restrict__ p, size_t n) {
  size_t i = (size_t)blockIdx.x * blockDim.x + threadIdx.x;
  size_t st = (size_t)gridDim.x * blockDim.x;
  for (; i < n; i += st) p[i] = 0.0f;
}

#define TILES_PER_BLOCK 16

// ---------------------------------------------------------------------------
// Edge kernel: per 16-edge tile, three fused WMMA GEMMs + SiLU, message
// aggregation via fp32 atomics (L2-resident), coord-weight dot + coord atomics.
// 256 threads = 8 waves; wave w owns output columns [w*16, w*16+16).
// ---------------------------------------------------------------------------
__global__ __launch_bounds__(256)
void egnn_edge_kernel(const float* __restrict__ features,
                      const float* __restrict__ coords,
                      const int*   __restrict__ eidx,
                      const float* __restrict__ Wm1,   // 257x128 (row 256 = dist)
                      const float* __restrict__ bm1,
                      const float* __restrict__ bm2,
                      const float* __restrict__ bc1,
                      const float* __restrict__ Wc2,   // 128x1
                      const float* __restrict__ bc2,
                      const unsigned short* __restrict__ pWm1, // 8x8 frags (K=256)
                      const unsigned short* __restrict__ pWm2, // 4x8 frags
                      const unsigned short* __restrict__ pWc1, // 4x8 frags
                      float* __restrict__ agg,
                      float* __restrict__ cacc,
                      int nEdges, int numTiles)
{
  __shared__ __align__(16) unsigned short sFeat[16][256]; // [feat(row)|feat(col)] bf16
  __shared__ __align__(16) unsigned short sH[16][128];    // layer-1 output bf16
  __shared__ __align__(16) unsigned short sM[16][128];    // messages bf16
  __shared__ __align__(16) float sT[16][128];             // layer-3 output fp32
  __shared__ float sDiff[16][3];
  __shared__ float sDist[16];
  __shared__ int   sRow[16];
  __shared__ int   sCol[16];
  __shared__ float sCW[16];

  const int t    = threadIdx.x;
  const int lane = t & 31;
  const int wave = t >> 5;                 // N-tile index 0..7
  const int nIdx = wave * 16 + (lane & 15);
  const int mHi  = (lane >= 16) ? 8 : 0;
  const int mRow = lane & 15;
  const int aSub = (lane >= 16) ? 8 : 0;   // per-lane K sub-offset for A frags

  const float biasM1 = bm1[nIdx];
  const float wLast  = Wm1[256 * 128 + nIdx];
  const float biasM2 = bm2[nIdx];
  const float biasC1 = bc1[nIdx];
  const float biasC2 = bc2[0];

  // hoist per-thread Wc2 chunk (8 coefficients) used by the coord-weight dot
  const int dotJ = t & 15;
  float4 wc2a = *(const float4*)(Wc2 + dotJ * 8);
  float4 wc2b = *(const float4*)(Wc2 + dotJ * 8 + 4);

  for (int it = 0; it < TILES_PER_BLOCK; ++it) {
    const int tile = blockIdx.x * TILES_PER_BLOCK + it;
    if (tile >= numTiles) break;           // uniform within block
    const int e0 = tile * 16;

    // ---- stage 0: indices, coord diff / dist --------------------------------
    if (t < 16) {
      int e  = e0 + t;
      int ec = (e < nEdges) ? e : 0;
      int r  = eidx[ec];
      int c  = eidx[nEdges + ec];
      sRow[t] = r; sCol[t] = c;
      float dx = coords[3 * r + 0] - coords[3 * c + 0];
      float dy = coords[3 * r + 1] - coords[3 * c + 1];
      float dz = coords[3 * r + 2] - coords[3 * c + 2];
      sDiff[t][0] = dx; sDiff[t][1] = dy; sDiff[t][2] = dz;
      sDist[t] = dx * dx + dy * dy + dz * dz;
      sCW[t] = 0.0f;
    }
    __syncthreads();

    // ---- gather features[row] / features[col] -> bf16 LDS -------------------
    {
      const int e = t >> 4;
      const int j = t & 15;
      {
        const float* src = features + (size_t)sRow[e] * 128 + j * 8;
        float4 a = *(const float4*)(src);
        float4 b = *(const float4*)(src + 4);
        uint4 p;
        p.x = (unsigned)f2bf(a.x) | ((unsigned)f2bf(a.y) << 16);
        p.y = (unsigned)f2bf(a.z) | ((unsigned)f2bf(a.w) << 16);
        p.z = (unsigned)f2bf(b.x) | ((unsigned)f2bf(b.y) << 16);
        p.w = (unsigned)f2bf(b.z) | ((unsigned)f2bf(b.w) << 16);
        *(uint4*)&sFeat[e][j * 8] = p;
      }
      {
        const float* src = features + (size_t)sCol[e] * 128 + j * 8;
        float4 a = *(const float4*)(src);
        float4 b = *(const float4*)(src + 4);
        uint4 p;
        p.x = (unsigned)f2bf(a.x) | ((unsigned)f2bf(a.y) << 16);
        p.y = (unsigned)f2bf(a.z) | ((unsigned)f2bf(a.w) << 16);
        p.z = (unsigned)f2bf(b.x) | ((unsigned)f2bf(b.y) << 16);
        p.w = (unsigned)f2bf(b.z) | ((unsigned)f2bf(b.w) << 16);
        *(uint4*)&sFeat[e][128 + j * 8] = p;
      }
    }
    __syncthreads();

    // ---- layer 1: h = silu([fr|fc|dist] @ Wm1 + bm1), K = 256 (+rank-1) ----
    v8f acc;
#pragma unroll
    for (int g = 0; g < 8; ++g)
      acc[g] = biasM1 + sDist[g + mHi] * wLast;    // bias + dist*Wm1[256,:]
#pragma unroll
    for (int kt = 0; kt < 8; ++kt) {
      ABFrag a, b;
      const unsigned short* ap = &sFeat[mRow][kt * 32 + aSub];
      a.q[0] = *(const uint4*)(ap);
      a.q[1] = *(const uint4*)(ap + 16);
      const uint4* bp = (const uint4*)(pWm1 + (((size_t)(kt * 8 + wave)) << 9) + lane * 16);
      b.q[0] = bp[0]; b.q[1] = bp[1];
      acc = __builtin_amdgcn_wmma_f32_16x16x32_bf16(false, a.v, false, b.v,
                                                    (short)0, acc, false, false);
    }
#pragma unroll
    for (int g = 0; g < 8; ++g)
      sH[g + mHi][nIdx] = f2bf(silu_f(acc[g]));
    __syncthreads();

    // ---- layer 2: messages = silu(h @ Wm2 + bm2); aggregate via atomics ----
#pragma unroll
    for (int g = 0; g < 8; ++g) acc[g] = biasM2;
#pragma unroll
    for (int kt = 0; kt < 4; ++kt) {
      ABFrag a, b;
      const unsigned short* ap = &sH[mRow][kt * 32 + aSub];
      a.q[0] = *(const uint4*)(ap);
      a.q[1] = *(const uint4*)(ap + 16);
      const uint4* bp = (const uint4*)(pWm2 + (((size_t)(kt * 8 + wave)) << 9) + lane * 16);
      b.q[0] = bp[0]; b.q[1] = bp[1];
      acc = __builtin_amdgcn_wmma_f32_16x16x32_bf16(false, a.v, false, b.v,
                                                    (short)0, acc, false, false);
    }
#pragma unroll
    for (int g = 0; g < 8; ++g) {
      float msg = silu_f(acc[g]);
      int M = g + mHi;
      sM[M][nIdx] = f2bf(msg);
      if (e0 + M < nEdges)
        atomicAdd(&agg[(size_t)sRow[M] * 128 + nIdx], msg);
    }
    __syncthreads();

    // ---- layer 3: t = silu(messages @ Wc1 + bc1) ----------------------------
#pragma unroll
    for (int g = 0; g < 8; ++g) acc[g] = biasC1;
#pragma unroll
    for (int kt = 0; kt < 4; ++kt) {
      ABFrag a, b;
      const unsigned short* ap = &sM[mRow][kt * 32 + aSub];
      a.q[0] = *(const uint4*)(ap);
      a.q[1] = *(const uint4*)(ap + 16);
      const uint4* bp = (const uint4*)(pWc1 + (((size_t)(kt * 8 + wave)) << 9) + lane * 16);
      b.q[0] = bp[0]; b.q[1] = bp[1];
      acc = __builtin_amdgcn_wmma_f32_16x16x32_bf16(false, a.v, false, b.v,
                                                    (short)0, acc, false, false);
    }
#pragma unroll
    for (int g = 0; g < 8; ++g)
      sT[g + mHi][nIdx] = silu_f(acc[g]);
    __syncthreads();

    // ---- cw = t @ Wc2 + bc2; coord update atomics ---------------------------
    {
      const int e = t >> 4;
      const float* row = &sT[e][dotJ * 8];
      float part = row[0] * wc2a.x + row[1] * wc2a.y + row[2] * wc2a.z +
                   row[3] * wc2a.w + row[4] * wc2b.x + row[5] * wc2b.y +
                   row[6] * wc2b.z + row[7] * wc2b.w;
      atomicAdd(&sCW[e], part);
    }
    __syncthreads();
    if (t < 16 && (e0 + t) < nEdges) {
      float cw = sCW[t] + biasC2;
      int r = sRow[t];
      atomicAdd(&cacc[3 * r + 0], cw * sDiff[t][0]);
      atomicAdd(&cacc[3 * r + 1], cw * sDiff[t][1]);
      atomicAdd(&cacc[3 * r + 2], cw * sDiff[t][2]);
    }
    __syncthreads();
  }
}

// ---------------------------------------------------------------------------
// Node kernel: new_features = silu([features|agg] @ Wf + bf), K = 256;
// new_coords = coords + cacc.
// ---------------------------------------------------------------------------
__global__ __launch_bounds__(256)
void egnn_node_kernel(const float* __restrict__ features,
                      const float* __restrict__ coords,
                      const float* __restrict__ agg,
                      const float* __restrict__ cacc,
                      const float* __restrict__ biasF,
                      const unsigned short* __restrict__ pWf,  // 8x8 frags (K=256)
                      float* __restrict__ outF,
                      float* __restrict__ outC,
                      int nNodes)
{
  __shared__ __align__(16) unsigned short sA[16][256];
  const int t    = threadIdx.x;
  const int lane = t & 31;
  const int wave = t >> 5;
  const int m0   = blockIdx.x * 16;

  // gather [features | aggregated] -> bf16 LDS
  {
    const int m = t >> 4;
    const int j = t & 15;
    int gn = m0 + m; if (gn >= nNodes) gn = nNodes - 1;
    {
      const float* src = features + (size_t)gn * 128 + j * 8;
      float4 a = *(const float4*)(src);
      float4 b = *(const float4*)(src + 4);
      uint4 p;
      p.x = (unsigned)f2bf(a.x) | ((unsigned)f2bf(a.y) << 16);
      p.y = (unsigned)f2bf(a.z) | ((unsigned)f2bf(a.w) << 16);
      p.z = (unsigned)f2bf(b.x) | ((unsigned)f2bf(b.y) << 16);
      p.w = (unsigned)f2bf(b.z) | ((unsigned)f2bf(b.w) << 16);
      *(uint4*)&sA[m][j * 8] = p;
    }
    {
      const float* src = agg + (size_t)gn * 128 + j * 8;
      float4 a = *(const float4*)(src);
      float4 b = *(const float4*)(src + 4);
      uint4 p;
      p.x = (unsigned)f2bf(a.x) | ((unsigned)f2bf(a.y) << 16);
      p.y = (unsigned)f2bf(a.z) | ((unsigned)f2bf(a.w) << 16);
      p.z = (unsigned)f2bf(b.x) | ((unsigned)f2bf(b.y) << 16);
      p.w = (unsigned)f2bf(b.z) | ((unsigned)f2bf(b.w) << 16);
      *(uint4*)&sA[m][128 + j * 8] = p;
    }
  }
  __syncthreads();

  const int nIdx = wave * 16 + (lane & 15);
  const int mHi  = (lane >= 16) ? 8 : 0;
  const int mRow = lane & 15;
  const int aSub = (lane >= 16) ? 8 : 0;

  v8f acc;
  const float bias = biasF[nIdx];
#pragma unroll
  for (int g = 0; g < 8; ++g) acc[g] = bias;
#pragma unroll
  for (int kt = 0; kt < 8; ++kt) {
    ABFrag a, b;
    const unsigned short* ap = &sA[mRow][kt * 32 + aSub];
    a.q[0] = *(const uint4*)(ap);
    a.q[1] = *(const uint4*)(ap + 16);
    const uint4* bp = (const uint4*)(pWf + (((size_t)(kt * 8 + wave)) << 9) + lane * 16);
    b.q[0] = bp[0]; b.q[1] = bp[1];
    acc = __builtin_amdgcn_wmma_f32_16x16x32_bf16(false, a.v, false, b.v,
                                                  (short)0, acc, false, false);
  }
#pragma unroll
  for (int g = 0; g < 8; ++g) {
    int gm = m0 + g + mHi;
    if (gm < nNodes)
      outF[(size_t)gm * 128 + nIdx] = silu_f(acc[g]);
  }

  if (t < 48) {
    int m = t / 3, d = t - 3 * m;
    int gn = m0 + m;
    if (gn < nNodes)
      outC[(size_t)gn * 3 + d] = coords[(size_t)gn * 3 + d] + cacc[(size_t)gn * 3 + d];
  }
}

// ---------------------------------------------------------------------------
// Launch
// ---------------------------------------------------------------------------
extern "C" void kernel_launch(void* const* d_in, const int* in_sizes, int n_in,
                              void* d_out, int out_size, void* d_ws, size_t ws_size,
                              hipStream_t stream) {
  const float* features = (const float*)d_in[0];
  const float* coords   = (const float*)d_in[1];
  const int*   eidx     = (const int*)d_in[2];
  const float* Wm1 = (const float*)d_in[3];
  const float* bm1 = (const float*)d_in[4];
  const float* Wm2 = (const float*)d_in[5];
  const float* bm2 = (const float*)d_in[6];
  const float* Wc1 = (const float*)d_in[7];
  const float* bc1 = (const float*)d_in[8];
  const float* Wc2 = (const float*)d_in[9];
  const float* bc2 = (const float*)d_in[10];
  const float* Wf  = (const float*)d_in[11];
  const float* bfv = (const float*)d_in[12];

  const int nNodes = in_sizes[0] / 128;
  const int nEdges = in_sizes[2] / 2;

  // workspace layout
  char* ws = (char*)d_ws;
  unsigned short* pWm1 = (unsigned short*)(ws);                     // 65536 B
  unsigned short* pWm2 = (unsigned short*)(ws + 65536);             // 32768 B
  unsigned short* pWc1 = (unsigned short*)(ws + 65536 + 32768);     // 32768 B
  unsigned short* pWf  = (unsigned short*)(ws + 131072);            // 65536 B
  float* agg  = (float*)(ws + 196608);                              // nNodes*128 f32
  float* cacc = agg + (size_t)nNodes * 128;                         // nNodes*3 f32

  // pack weights into WMMA B-fragment layout (bf16)
  pack_wmma_b<<<(32768 + 255) / 256, 256, 0, stream>>>(Wm1, 256, pWm1); // rows 0..255
  pack_wmma_b<<<(16384 + 255) / 256, 256, 0, stream>>>(Wm2, 128, pWm2);
  pack_wmma_b<<<(16384 + 255) / 256, 256, 0, stream>>>(Wc1, 128, pWc1);
  pack_wmma_b<<<(32768 + 255) / 256, 256, 0, stream>>>(Wf,  256, pWf);

  // zero aggregation buffers (agg and cacc are contiguous: 128+3 per node)
  zero_f32<<<2048, 256, 0, stream>>>(agg, (size_t)nNodes * 131);

  const int numTiles = (nEdges + 15) / 16;
  const int blocks = (numTiles + TILES_PER_BLOCK - 1) / TILES_PER_BLOCK;
  egnn_edge_kernel<<<blocks, 256, 0, stream>>>(
      features, coords, eidx, Wm1, bm1, bm2, bc1, Wc2, bc2,
      pWm1, pWm2, pWc1, agg, cacc, nEdges, numTiles);

  float* outF = (float*)d_out;
  float* outC = outF + (size_t)nNodes * 128;
  egnn_node_kernel<<<(nNodes + 15) / 16, 256, 0, stream>>>(
      features, coords, agg, cacc, bfv, pWf, outF, outC, nNodes);
}
